// NeZhaSelfAttention_38070590112307
// MI455X (gfx1250) — matrix operand order repaired and checked by reading, not compiled, not run on hardware
//
#include <hip/hip_runtime.h>

// ---------------------------------------------------------------------------
// NeZha self-attention for MI455X (gfx1250, wave32, WMMA 16x16x32 bf16)
// B=8, S=512, H=1024, NH=16, HD=64, MAX_REL=64
// - All WMMA B operands stored [N][K] row-major -> fragments are contiguous
//   16B runs (ds_load_b128).
// - Global->LDS staging uses gfx1250 async copies
//   (global_load_async_to_lds_b128, ASYNCcnt); qkv GEMM double-buffers its
//   K-loop (peeled tail => branch-free steady state, no accumulator copies).
// ---------------------------------------------------------------------------

typedef __attribute__((ext_vector_type(16))) __bf16 bf16x16;
typedef __attribute__((ext_vector_type(8)))  float  f32x8;

#define Bsz   8
#define Ssz   512
#define Hsz   1024
#define NHh   16
#define HDd   64
#define BHc   128   // B*NH
#define RVv   129   // relative vocab (2*64+1)
#define RPp   160   // vocab padded to multiple of 32 (WMMA K)

__device__ __forceinline__ f32x8 zero8() {
  f32x8 z = {0.f,0.f,0.f,0.f,0.f,0.f,0.f,0.f};
  return z;
}

__device__ __forceinline__ f32x8 wmma_bf(bf16x16 a, bf16x16 b, f32x8 c) {
  return __builtin_amdgcn_wmma_f32_16x16x32_bf16(false, a, false, b, (short)0, c,
                                                 false, false);
}

// ---- gfx1250 async global->LDS copies (tracked by ASYNCcnt) ----------------
__device__ __forceinline__ void async_cp128(void* lds_dst, const void* gsrc) {
  unsigned l = (unsigned)(size_t)lds_dst;
  asm volatile("global_load_async_to_lds_b128 %0, %1, off"
               :: "v"(l), "v"(gsrc) : "memory");
}
__device__ __forceinline__ void async_cp32(void* lds_dst, const void* gsrc) {
  unsigned l = (unsigned)(size_t)lds_dst;
  asm volatile("global_load_async_to_lds_b32 %0, %1, off"
               :: "v"(l), "v"(gsrc) : "memory");
}
__device__ __forceinline__ void wait_async0() {
  asm volatile("s_wait_asynccnt 0x0" ::: "memory");
}

// A fragment: 16x32 bf16 from row-major [M][K] tile (ld = K stride).
// ISA: vgpr v holds K=(v>>2)*16 + hi*8 + (v&3)*2 (+1)  =>  two contiguous
// 8-element runs per lane: [hi*8 .. hi*8+7] and [16+hi*8 .. 16+hi*8+7].
__device__ __forceinline__ bf16x16 ldA_bf(const __bf16* s, int ld, int lane) {
  int row = lane & 15, hi = lane >> 4;
  union { bf16x16 v; uint4 q[2]; } a;
  a.q[0] = *(const uint4*)(s + row * ld + (hi << 3));
  a.q[1] = *(const uint4*)(s + row * ld + 16 + (hi << 3));
  return a.v;
}

// Same A fragment from fp32 source (vector loads + inline bf16 convert).
__device__ __forceinline__ bf16x16 ldA_f32(const float* s, int ld, int lane) {
  int row = lane & 15, hi = lane >> 4;
  float4 f0 = *(const float4*)(s + row * ld + (hi << 3));
  float4 f1 = *(const float4*)(s + row * ld + (hi << 3) + 4);
  float4 f2 = *(const float4*)(s + row * ld + 16 + (hi << 3));
  float4 f3 = *(const float4*)(s + row * ld + 16 + (hi << 3) + 4);
  bf16x16 a;
  a[0] = (__bf16)f0.x;  a[1] = (__bf16)f0.y;  a[2]  = (__bf16)f0.z;  a[3]  = (__bf16)f0.w;
  a[4] = (__bf16)f1.x;  a[5] = (__bf16)f1.y;  a[6]  = (__bf16)f1.z;  a[7]  = (__bf16)f1.w;
  a[8] = (__bf16)f2.x;  a[9] = (__bf16)f2.y;  a[10] = (__bf16)f2.z;  a[11] = (__bf16)f2.w;
  a[12] = (__bf16)f3.x; a[13] = (__bf16)f3.y; a[14] = (__bf16)f3.z;  a[15] = (__bf16)f3.w;
  return a;
}

// B fragment: 32x16 (KxN) gathered from an [N][K] row-major tile (ld = K stride).
// Same K order as A: two contiguous 8-element runs per lane at
// s[col*ld + hi*8] and s[col*ld + 16 + hi*8], col=lane&15.
__device__ __forceinline__ bf16x16 ldB_bf(const __bf16* s, int ld, int lane) {
  int col = lane & 15, hi = lane >> 4;
  union { bf16x16 v; uint4 q[2]; } b;
  b.q[0] = *(const uint4*)(s + col * ld + (hi << 3));
  b.q[1] = *(const uint4*)(s + col * ld + 16 + (hi << 3));
  return b.v;
}

// ---------------------------------------------------------------------------
// Prep kernels: fp32 -> bf16 (pure convert, no transposes needed)
// ---------------------------------------------------------------------------
__global__ void cvt_bf16(const float* __restrict__ x, __bf16* __restrict__ y) {
  int i = (blockIdx.x * 256 + threadIdx.x) * 4;
  float4 f = *(const float4*)(x + i);
  y[i+0] = (__bf16)f.x; y[i+1] = (__bf16)f.y;
  y[i+2] = (__bf16)f.z; y[i+3] = (__bf16)f.w;
}

__global__ void build_tab(const float* __restrict__ rel, __bf16* __restrict__ tab,
                          __bf16* __restrict__ tabT) {
  int i = blockIdx.x * 256 + threadIdx.x;          // 160*64
  if (i >= RPp * HDd) return;
  int r = i >> 6, d = i & 63;
  // tab[r,:] == rel_pos[f=64, t=r, :] exactly (clip(r-64,±64)+64 == r, r in [0,128])
  float v = (r < RVv) ? rel[((size_t)(64 * Ssz + r)) * HDd + d] : 0.f;
  tab[r * HDd + d]  = (__bf16)v;   // [r][d] : B operand for RQ = Q x tab^T
  tabT[d * RPp + r] = (__bf16)v;   // [d][r] : B operand for prow x tab
}

// ---------------------------------------------------------------------------
// QKV projection: [4096x1024] x W^T + bias. B operand = W natural [n][k].
// Double-buffered K-loop with peeled tail: async-stage next tile while WMMA
// runs on current. Q,K stored [bh][s][d]; V stored transposed [bh][d][s].
// ---------------------------------------------------------------------------
__global__ __launch_bounds__(256) void qkv_gemm(
    const __bf16* __restrict__ Xb, const __bf16* __restrict__ Wc_all,
    const float* __restrict__ bq, const float* __restrict__ bk,
    const float* __restrict__ bv,
    __bf16* __restrict__ Qg, __bf16* __restrict__ Kg, __bf16* __restrict__ Vtg) {
  const int n0 = blockIdx.x * 128;
  const int m0 = blockIdx.y * 128;
  const int mode = blockIdx.z;
  const __bf16* Wc = Wc_all + (size_t)mode * Hsz * Hsz;
  const float* bias = (mode == 0) ? bq : (mode == 1) ? bk : bv;

  const int tid = threadIdx.x, lane = tid & 31, wave = tid >> 5;
  const int wm = wave >> 1, wn = wave & 1;
  const int colL = lane & 15, hi = lane >> 4;

  __shared__ alignas(16) __bf16 sA[2][128 * 32];   // [m][k], double-buffered
  __shared__ alignas(16) __bf16 sB[2][128 * 32];   // [n][k], double-buffered

  const int r4 = tid >> 2, c4 = tid & 3;           // 256 threads: row, uint4-col

  auto stage = [&](int buf, int k0) {
    const uint4* xs = (const uint4*)(Xb + (size_t)m0 * Hsz + k0);
    const uint4* bs = (const uint4*)(Wc + (size_t)n0 * Hsz + k0);
    uint4* ad = (uint4*)sA[buf];
    uint4* bd = (uint4*)sB[buf];
    // 512 uint4 per tile / 256 threads = 2 each; global row stride = 128 uint4
    async_cp128(ad + tid,       xs + r4 * 128 + c4);
    async_cp128(ad + tid + 256, xs + (r4 + 64) * 128 + c4);
    async_cp128(bd + tid,       bs + r4 * 128 + c4);
    async_cp128(bd + tid + 256, bs + (r4 + 64) * 128 + c4);
  };

  f32x8 acc[2][4];
#pragma unroll
  for (int mt = 0; mt < 2; ++mt)
#pragma unroll
    for (int nt = 0; nt < 4; ++nt) acc[mt][nt] = zero8();

  auto compute = [&](int cur) {
    bf16x16 a0 = ldA_bf(sA[cur] + (wm * 32) * 32, 32, lane);
    bf16x16 a1 = ldA_bf(sA[cur] + (wm * 32 + 16) * 32, 32, lane);
#pragma unroll
    for (int nt = 0; nt < 4; ++nt) {
      bf16x16 bfr = ldB_bf(sB[cur] + (wn * 64 + nt * 16) * 32, 32, lane);
      acc[0][nt] = wmma_bf(a0, bfr, acc[0][nt]);
      acc[1][nt] = wmma_bf(a1, bfr, acc[1][nt]);
    }
  };

  stage(0, 0);
  wait_async0();
  __syncthreads();

  // Steady state: branch-free body (stage next, compute current, wait, barrier).
#pragma unroll 1
  for (int step = 0; step < 31; ++step) {
    const int cur = step & 1;
    stage(cur ^ 1, (step + 1) * 32);
    compute(cur);
    wait_async0();
    __syncthreads();
  }
  compute(1);  // peeled final K-step (31 & 1)

#pragma unroll
  for (int nt = 0; nt < 4; ++nt) {
    int n = n0 + wn * 64 + nt * 16 + colL;
    float bval = bias[n];
#pragma unroll
    for (int mt = 0; mt < 2; ++mt)
#pragma unroll
      for (int v = 0; v < 8; ++v) {
        int m = m0 + wm * 32 + mt * 16 + v + (hi << 3);
        __bf16 val = (__bf16)(acc[mt][nt][v] + bval);
        int b = m >> 9, s = m & 511, h = n >> 6, d = n & 63;
        int bh = b * NHh + h;
        if (mode == 0)      Qg[((size_t)bh * Ssz + s) * HDd + d] = val;
        else if (mode == 1) Kg[((size_t)bh * Ssz + s) * HDd + d] = val;
        else                Vtg[((size_t)bh * HDd + d) * Ssz + s] = val;
      }
  }
}

// ---------------------------------------------------------------------------
// RQ = Q x tab^T : fp32 [bh, 512, 160].  B operand = tab [r][d].
// ---------------------------------------------------------------------------
__global__ __launch_bounds__(256) void rq_gemm(const __bf16* __restrict__ Qg,
                                               const __bf16* __restrict__ tab,
                                               float* __restrict__ RQ) {
  const int f0 = blockIdx.x * 128;
  const int bh = blockIdx.y;
  const int tid = threadIdx.x, lane = tid & 31, wave = tid >> 5;
  const int colL = lane & 15, hi = lane >> 4;

  __shared__ alignas(16) __bf16 sA[128 * HDd];   // [f][d]   16 KB
  __shared__ alignas(16) __bf16 sB[RPp * HDd];   // [r][d]   20 KB

  {
    const uint4* src = (const uint4*)(Qg + ((size_t)bh * Ssz + f0) * HDd);
    uint4* dst = (uint4*)sA;
#pragma unroll
    for (int j = 0; j < 4; ++j) {
      int i = tid + j * 256;
      async_cp128(dst + i, src + i);
    }
    const uint4* ts = (const uint4*)tab;
    uint4* td = (uint4*)sB;
#pragma unroll
    for (int j = 0; j < 5; ++j) {
      int i = tid + j * 256;
      async_cp128(td + i, ts + i);
    }
  }
  wait_async0();
  __syncthreads();

  f32x8 acc[10];
#pragma unroll
  for (int nt = 0; nt < 10; ++nt) acc[nt] = zero8();

#pragma unroll
  for (int kk = 0; kk < HDd; kk += 32) {
    bf16x16 a = ldA_bf(sA + (wave * 16) * HDd + kk, HDd, lane);
#pragma unroll
    for (int nt = 0; nt < 10; ++nt) {
      bf16x16 bfr = ldB_bf(sB + (nt * 16) * HDd + kk, HDd, lane);
      acc[nt] = wmma_bf(a, bfr, acc[nt]);
    }
  }

#pragma unroll
  for (int nt = 0; nt < 10; ++nt)
#pragma unroll
    for (int v = 0; v < 8; ++v) {
      int row = wave * 16 + v + (hi << 3);
      RQ[((size_t)bh * Ssz + f0 + row) * RPp + nt * 16 + colL] = acc[nt][v];
    }
}

// ---------------------------------------------------------------------------
// Attention: flash-style online softmax + rel-pos via RQ gather / prow scatter.
// Grid (8 f-blocks of 64 rows, 128 bh), 128 threads = 4 waves; wave w owns
// rows [w*16, w*16+16) exclusively -> softmax state needs no cross-wave sync.
// LDS ~157 KB -> 2 workgroups per 320 KB WGP.
// ---------------------------------------------------------------------------
__global__ __launch_bounds__(128) void attn_kernel(
    const __bf16* __restrict__ Qg, const __bf16* __restrict__ Kg,
    const __bf16* __restrict__ Vtg, const __bf16* __restrict__ tabT,
    const float* __restrict__ RQ, const float* __restrict__ mask,
    float* __restrict__ out) {
  const int f0 = blockIdx.x * 64;
  const int bh = blockIdx.y;
  const int bb = bh >> 4, hh = bh & 15;
  const int tid = threadIdx.x, lane = tid & 31, wave = tid >> 5;
  const int colL = lane & 15, hi = lane >> 4;

  __shared__ alignas(16) __bf16 sQ[64 * HDd];      //  8 KB  [f][d]
  __shared__ alignas(16) __bf16 sK[128 * HDd];     // 16 KB  [t][d]  (B of QK^T)
  __shared__ alignas(16) __bf16 sVt[HDd * 128];    // 16 KB  [d][t]  (B of P.V)
  __shared__ alignas(16) __bf16 sP[64 * 128];      // 16 KB  [f][t]  (A of P.V)
  __shared__ alignas(16) float  sRQ[64 * RPp];     // 40 KB
  __shared__ alignas(16) float  sProw[64 * RPp];   // 40 KB
  __shared__ alignas(16) __bf16 sTabT[HDd * RPp];  // 20 KB  [d][r]  (B of prow.tab)
  __shared__ float sMask[128];
  __shared__ float sM[64], sL[64], sF[64];

  {  // async-stage Q block, tabT, RQ rows; init stats + prow
    const uint4* src = (const uint4*)(Qg + ((size_t)bh * Ssz + f0) * HDd);
    uint4* dst = (uint4*)sQ;
#pragma unroll
    for (int j = 0; j < 4; ++j) {
      int i = tid + j * 128;
      async_cp128(dst + i, src + i);
    }
    const uint4* ts = (const uint4*)tabT;
    uint4* td = (uint4*)sTabT;
#pragma unroll
    for (int j = 0; j < 10; ++j) {
      int i = tid + j * 128;
      async_cp128(td + i, ts + i);
    }
    const uint4* rs = (const uint4*)(RQ + ((size_t)bh * Ssz + f0) * RPp);
    uint4* rd = (uint4*)sRQ;
#pragma unroll
    for (int j = 0; j < 20; ++j) {
      int i = tid + j * 128;
      async_cp128(rd + i, rs + i);
    }
  }
  if (tid < 64) { sM[tid] = -3.0e38f; sL[tid] = 0.f; }
  for (int i = tid; i < 64 * RPp; i += 128) sProw[i] = 0.f;

  f32x8 o[4];
#pragma unroll
  for (int dt = 0; dt < 4; ++dt) o[dt] = zero8();

  for (int tb = 0; tb < 4; ++tb) {
    const int t0 = tb * 128;
    __syncthreads();  // all waves done reading previous sK/sVt
    {  // async-stage K tile (linear), V^T tile (strided rows), mask slice
      const uint4* ks = (const uint4*)(Kg + ((size_t)bh * Ssz + t0) * HDd);
      uint4* kd = (uint4*)sK;
#pragma unroll
      for (int j = 0; j < 8; ++j) {
        int i = tid + j * 128;
        async_cp128(kd + i, ks + i);
      }
      const uint4* vs = (const uint4*)(Vtg + (size_t)bh * HDd * Ssz + t0);
      uint4* vd = (uint4*)sVt;
#pragma unroll
      for (int j = 0; j < 8; ++j) {
        int i = tid + j * 128;
        int r = i >> 4, c = i & 15;
        async_cp128(vd + i, vs + r * 64 + c);  // global row stride = 64 uint4
      }
      async_cp32(&sMask[tid], &mask[bb * Ssz + t0 + tid]);
    }
    wait_async0();
    __syncthreads();

    // ---- scores = Q x K^T (WMMA) ----
    f32x8 sc[8];
#pragma unroll
    for (int nt = 0; nt < 8; ++nt) sc[nt] = zero8();
#pragma unroll
    for (int kk = 0; kk < HDd; kk += 32) {
      bf16x16 a = ldA_bf(sQ + (wave * 16) * HDd + kk, HDd, lane);
#pragma unroll
      for (int nt = 0; nt < 8; ++nt) {
        bf16x16 bfr = ldB_bf(sK + (nt * 16) * HDd + kk, HDd, lane);
        sc[nt] = wmma_bf(a, bfr, sc[nt]);
      }
    }

    // ---- + rel (sRQ gather), *1/sqrt(HD), + mask ----
#pragma unroll
    for (int nt = 0; nt < 8; ++nt)
#pragma unroll
      for (int v = 0; v < 8; ++v) {
        int row = wave * 16 + v + (hi << 3);
        int fg = f0 + row;
        int tl = nt * 16 + colL;
        int rr = (t0 + tl) - fg;
        rr = (rr < -64) ? -64 : (rr > 64 ? 64 : rr);
        float rel = sRQ[row * RPp + (rr + 64)];
        sc[nt][v] = (sc[nt][v] + rel) * 0.125f + sMask[tl];
      }

    // ---- online softmax: row max (16-lane half reduction) ----
    float mx[8];
#pragma unroll
    for (int v = 0; v < 8; ++v) {
      float m = sc[0][v];
#pragma unroll
      for (int nt = 1; nt < 8; ++nt) m = fmaxf(m, sc[nt][v]);
      m = fmaxf(m, __shfl_xor(m, 1, 32));
      m = fmaxf(m, __shfl_xor(m, 2, 32));
      m = fmaxf(m, __shfl_xor(m, 4, 32));
      m = fmaxf(m, __shfl_xor(m, 8, 32));
      mx[v] = m;
    }
    if (colL < 8) {
      float rmax = mx[0];
#pragma unroll
      for (int v = 1; v < 8; ++v)
        if ((lane & 7) == v) rmax = mx[v];
      int row = wave * 16 + (lane & 7) + (hi << 3);
      float mo = sM[row];
      float mn = fmaxf(mo, rmax);
      sF[row] = __expf(mo - mn);
      sM[row] = mn;
    }

    // ---- p = exp(s - m); row sums; stash p (bf16) for P x V ----
    float mrow[8];
#pragma unroll
    for (int v = 0; v < 8; ++v) mrow[v] = sM[wave * 16 + v + (hi << 3)];
    float sum[8];
#pragma unroll
    for (int v = 0; v < 8; ++v) sum[v] = 0.f;
#pragma unroll
    for (int nt = 0; nt < 8; ++nt)
#pragma unroll
      for (int v = 0; v < 8; ++v) {
        float p = __expf(sc[nt][v] - mrow[v]);
        sum[v] += p;
        sc[nt][v] = p;
        sP[(wave * 16 + v + (hi << 3)) * 128 + nt * 16 + colL] = (__bf16)p;
      }
#pragma unroll
    for (int v = 0; v < 8; ++v) {
      float s = sum[v];
      s += __shfl_xor(s, 1, 32);
      s += __shfl_xor(s, 2, 32);
      s += __shfl_xor(s, 4, 32);
      s += __shfl_xor(s, 8, 32);
      sum[v] = s;
    }
    if (colL < 8) {
      float rsum = sum[0];
#pragma unroll
      for (int v = 1; v < 8; ++v)
        if ((lane & 7) == v) rsum = sum[v];
      int row = wave * 16 + (lane & 7) + (hi << 3);
      sL[row] = sL[row] * sF[row] + rsum;
    }

    // ---- rescale prow (vectorized), then scatter-add p by clipped distance ----
    for (int r16 = 0; r16 < 16; ++r16) {
      int row = wave * 16 + r16;
      float f = sF[row];
      float4* pr = (float4*)(sProw + row * RPp);
      for (int j = lane; j < RPp / 4; j += 32) {
        float4 x = pr[j];
        x.x *= f; x.y *= f; x.z *= f; x.w *= f;
        pr[j] = x;
      }
    }
#pragma unroll
    for (int nt = 0; nt < 8; ++nt)
#pragma unroll
      for (int v = 0; v < 8; ++v) {
        int row = wave * 16 + v + (hi << 3);
        int rr = (t0 + nt * 16 + colL) - (f0 + row);
        rr = (rr < -64) ? -64 : (rr > 64 ? 64 : rr);
        atomicAdd(&sProw[row * RPp + (rr + 64)], sc[nt][v]);
      }

    // ---- O *= factor; O += P x V (WMMA) ----
    float fct[8];
#pragma unroll
    for (int v = 0; v < 8; ++v) fct[v] = sF[wave * 16 + v + (hi << 3)];
#pragma unroll
    for (int dt = 0; dt < 4; ++dt)
#pragma unroll
      for (int v = 0; v < 8; ++v) o[dt][v] *= fct[v];
#pragma unroll
    for (int kk = 0; kk < 128; kk += 32) {
      bf16x16 a = ldA_bf(sP + (wave * 16) * 128 + kk, 128, lane);
#pragma unroll
      for (int dt = 0; dt < 4; ++dt) {
        bf16x16 bfr = ldB_bf(sVt + (dt * 16) * 128 + kk, 128, lane);
        o[dt] = wmma_bf(a, bfr, o[dt]);
      }
    }
  }

  // ---- O += prow x tab (WMMA over padded K=160) ----
#pragma unroll
  for (int kk = 0; kk < RPp; kk += 32) {
    bf16x16 a = ldA_f32(sProw + (wave * 16) * RPp + kk, RPp, lane);
#pragma unroll
    for (int dt = 0; dt < 4; ++dt) {
      bf16x16 bfr = ldB_bf(sTabT + (dt * 16) * RPp + kk, RPp, lane);
      o[dt] = wmma_bf(a, bfr, o[dt]);
    }
  }

  // ---- normalize by l and store: out[b, f, h*64 + d] ----
  float inv[8];
#pragma unroll
  for (int v = 0; v < 8; ++v) inv[v] = 1.0f / sL[wave * 16 + v + (hi << 3)];
#pragma unroll
  for (int dt = 0; dt < 4; ++dt)
#pragma unroll
    for (int v = 0; v < 8; ++v) {
      int row = wave * 16 + v + (hi << 3);
      int fg = f0 + row;
      int dg = dt * 16 + colL;
      out[((size_t)bb * Ssz + fg) * Hsz + hh * HDd + dg] = o[dt][v] * inv[v];
    }
}

// ---------------------------------------------------------------------------
extern "C" void kernel_launch(void* const* d_in, const int* in_sizes, int n_in,
                              void* d_out, int out_size, void* d_ws, size_t ws_size,
                              hipStream_t stream) {
  const float* hs   = (const float*)d_in[0];
  const float* mask = (const float*)d_in[1];
  const float* Wq   = (const float*)d_in[2];
  const float* bq   = (const float*)d_in[3];
  const float* Wk   = (const float*)d_in[4];
  const float* bk   = (const float*)d_in[5];
  const float* Wv   = (const float*)d_in[6];
  const float* bv   = (const float*)d_in[7];
  const float* rel  = (const float*)d_in[8];
  float* out = (float*)d_out;

  char* ws = (char*)d_ws;
  size_t off = 0;
  auto walloc = [&](size_t bytes) -> void* {
    void* p = ws + off;
    off += (bytes + 255) & ~(size_t)255;
    return p;
  };
  __bf16* Xb   = (__bf16*)walloc((size_t)Bsz * Ssz * Hsz * 2);        //  8 MB
  __bf16* Wc   = (__bf16*)walloc((size_t)3 * Hsz * Hsz * 2);          //  6 MB
  __bf16* Qg   = (__bf16*)walloc((size_t)BHc * Ssz * HDd * 2);        //  8 MB
  __bf16* Kg   = (__bf16*)walloc((size_t)BHc * Ssz * HDd * 2);        //  8 MB
  __bf16* Vtg  = (__bf16*)walloc((size_t)BHc * HDd * Ssz * 2);        //  8 MB
  __bf16* tab  = (__bf16*)walloc((size_t)RPp * HDd * 2);
  __bf16* tabT = (__bf16*)walloc((size_t)HDd * RPp * 2);
  float*  RQ   = (float*)walloc((size_t)BHc * Ssz * RPp * 4);         // 40 MB

  cvt_bf16<<<dim3((Bsz * Ssz * Hsz) / 1024), 256, 0, stream>>>(hs, Xb);
  cvt_bf16<<<dim3((Hsz * Hsz) / 1024), 256, 0, stream>>>(Wq, Wc + 0 * (size_t)Hsz * Hsz);
  cvt_bf16<<<dim3((Hsz * Hsz) / 1024), 256, 0, stream>>>(Wk, Wc + 1 * (size_t)Hsz * Hsz);
  cvt_bf16<<<dim3((Hsz * Hsz) / 1024), 256, 0, stream>>>(Wv, Wc + 2 * (size_t)Hsz * Hsz);
  build_tab<<<dim3((RPp * HDd + 255) / 256), 256, 0, stream>>>(rel, tab, tabT);

  qkv_gemm<<<dim3(Hsz / 128, (Bsz * Ssz) / 128, 3), 256, 0, stream>>>(
      Xb, Wc, bq, bk, bv, Qg, Kg, Vtg);

  rq_gemm<<<dim3(Ssz / 128, BHc), 256, 0, stream>>>(Qg, tab, RQ);

  attn_kernel<<<dim3(Ssz / 64, BHc), 128, 0, stream>>>(Qg, Kg, Vtg, tabT, RQ,
                                                       mask, out);
}
